// RecurrentSNN_910533066907
// MI455X (gfx1250) — matrix operand order repaired and checked by reading, not compiled, not run on hardware
//
#include <hip/hip_runtime.h>
#include <hip/hip_bf16.h>
#include <stddef.h>

// ---------------------------------------------------------------------------
// Recurrent spiking NN for MI455X (gfx1250, wave32, WMMA bf16 16x16x32).
// Phase 1: cur1_ff[t,b,h] = x[b,t,:] @ W1[h,:] + b1[h]   (bf16 WMMA GEMM -> ws)
// Phase 2: per-batch-tile time loop; Wrec/W2 staged in LDS as bf16,
//          spk1 kept in LDS, mem1/mem2 in WMMA accumulator registers.
// ---------------------------------------------------------------------------

typedef __bf16 bf16_t;
typedef __attribute__((ext_vector_type(16))) __bf16 v16bf;
typedef __attribute__((ext_vector_type(8)))  float  v8f;
typedef __attribute__((ext_vector_type(4)))  float  v4f;

#define N_IN   700
#define N_HID  256
#define N_CLS  20
#define N_T    100
#define N_B    1024
#define BETA   0.9f
#define THRESH 1.0f
#define OUT_TOTAL_IDX (N_T * N_B * N_CLS)   // 2,048,000

// ---- fragment fill helpers -------------------------------------------------
// 16-bit A-fragment (16x32): lane holds two contiguous 8-element K runs at
// k = 8*half and k = 16 + 8*half.  B-fragment (32x16): lane n holds the same
// two 8-element K runs.  (ISA 7.12.2 wave32 layouts.)

template <bool NT>
__device__ __forceinline__ v4f ld4(const float* p) {
  if (NT) return __builtin_nontemporal_load((const v4f*)p);
  return *(const v4f*)p;
}

// Convert 8 contiguous, 16B-aligned floats -> fragment elements [eb, eb+8).
template <bool NT>
__device__ __forceinline__ void cvt8_to(const float* p, v16bf& d, int eb) {
  v4f a = ld4<NT>(p);
  v4f b = ld4<NT>(p + 4);
#pragma unroll
  for (int j = 0; j < 4; ++j) {
    d[eb + j]     = (bf16_t)a[j];
    d[eb + 4 + j] = (bf16_t)b[j];
  }
}

// Tail version: clamp address in-bounds, zero the value by select (no branch).
__device__ __forceinline__ void cvt8_tail(const float* row, int kbase,
                                          v16bf& d, int eb) {
#pragma unroll
  for (int j = 0; j < 8; ++j) {
    int  k  = kbase + j;
    bool ok = (k < N_IN);
    float f = row[ok ? k : (N_IN - 1)];
    d[eb + j] = (bf16_t)(ok ? f : 0.f);
  }
}

// ---------------------------------------------------------------------------
// Phase 1: FF GEMM.  Block = 64 rows x 256 cols, 8 waves (2 row-groups x 4
// col-groups); wave = 32 rows x 64 cols = 2x4 WMMA tiles.  K = 700 as
// 21 clean chunks of 32 + 1 guarded tail chunk.
// ---------------------------------------------------------------------------
__global__ void __launch_bounds__(256)
snn_ff_gemm(const float* __restrict__ x, const float* __restrict__ W1,
            const float* __restrict__ b1, bf16_t* __restrict__ ff) {
  const int lane = threadIdx.x & 31;
  const int wave = threadIdx.x >> 5;
  const int half = lane >> 4;
  const int ln   = lane & 15;

  const int rowBase = blockIdx.x * 64 + (wave & 1) * 32;   // m base (2 subtiles)
  const int colBase = (wave >> 1) * 64;                    // 4 col tiles

  v8f acc[2][4];
#pragma unroll
  for (int s = 0; s < 2; ++s)
#pragma unroll
    for (int c = 0; c < 4; ++c)
      acc[s][c] = (v8f){0.f,0.f,0.f,0.f,0.f,0.f,0.f,0.f};

  // Row -> (t, b): m = t*1024 + b ; x is [b][t][i]
  const float* xrow[2];
#pragma unroll
  for (int s = 0; s < 2; ++s) {
    int m = rowBase + s * 16 + ln;
    int t = m >> 10;
    int b = m & 1023;
    xrow[s] = x + ((size_t)b * N_T + t) * N_IN;
  }
  const float* wrow[4];
#pragma unroll
  for (int c = 0; c < 4; ++c)
    wrow[c] = W1 + (size_t)(colBase + c * 16 + ln) * N_IN;

  // ---- 21 unguarded K chunks ----
  for (int kc = 0; kc < 21; ++kc) {
    const int kb = kc * 32;

    v16bf A[2];
#pragma unroll
    for (int s = 0; s < 2; ++s) {
      const float* ar = xrow[s] + kb + 8 * half;
      __builtin_prefetch(xrow[s] + kb + 64, 0, 1);   // global_prefetch_b8
      cvt8_to<true>(ar,      A[s], 0);
      cvt8_to<true>(ar + 16, A[s], 8);
    }

#pragma unroll
    for (int c = 0; c < 4; ++c) {
      const float* br = wrow[c] + kb + 8 * half;
      v16bf Bf;
      cvt8_to<false>(br,      Bf, 0);
      cvt8_to<false>(br + 16, Bf, 8);
#pragma unroll
      for (int s = 0; s < 2; ++s) {
        acc[s][c] = __builtin_amdgcn_wmma_f32_16x16x32_bf16(
            false, A[s], false, Bf, (short)0, acc[s][c], false, false);
      }
    }
  }

  // ---- tail chunk: kb = 672, valid k < 700 (select-guarded, branch-free) ----
  {
    const int kb = 21 * 32;   // 672
    v16bf A[2];
#pragma unroll
    for (int s = 0; s < 2; ++s) {
      // run0: k = 672+8h .. 679+8h  -> always valid (max 687)
      cvt8_to<true>(xrow[s] + kb + 8 * half, A[s], 0);
      // run1: k = 688+8h .. 695+8h  -> can cross 700 when h = 1
      cvt8_tail(xrow[s], kb + 16 + 8 * half, A[s], 8);
    }
#pragma unroll
    for (int c = 0; c < 4; ++c) {
      v16bf Bf;
      cvt8_tail(wrow[c], kb + 8 * half,      Bf, 0);
      cvt8_tail(wrow[c], kb + 16 + 8 * half, Bf, 8);
#pragma unroll
      for (int s = 0; s < 2; ++s) {
        acc[s][c] = __builtin_amdgcn_wmma_f32_16x16x32_bf16(
            false, A[s], false, Bf, (short)0, acc[s][c], false, false);
      }
    }
  }

  // Epilogue: + b1, store bf16 currents to ws as [t*B + b][h] == [t][b][h].
#pragma unroll
  for (int c = 0; c < 4; ++c) {
    const int n = colBase + c * 16 + ln;
    const float bias = b1[n];
#pragma unroll
    for (int s = 0; s < 2; ++s) {
#pragma unroll
      for (int r = 0; r < 8; ++r) {
        int m = rowBase + s * 16 + r + 8 * half;
        __builtin_nontemporal_store((bf16_t)(acc[s][c][r] + bias),
                                    ff + (size_t)m * N_HID + n);
      }
    }
  }
}

// ---------------------------------------------------------------------------
// Phase 2: recurrent time loop.  64 blocks x 256 threads; each block owns
// batch rows [b0, b0+16).  LDS (dynamic, ~157 KB):
//   sWb  [256][264] bf16  : Wrec rows (B-matrix, padded stride vs bank conflicts)
//   sW2  [ 32][264] bf16  : W2 rows padded to 32 cols (cols >= 20 zero)
//   sSpk [ 16][264] bf16  : spk1 tile (A-matrix for both matmuls)
//   sRed [1] float        : block spike-count reduction
// Waves 0..7 each own hidden cols [w*32, w*32+32); waves 0..1 additionally
// run the (20 -> padded 32)-wide output layer.
// ---------------------------------------------------------------------------
__global__ void __launch_bounds__(256)
snn_recurrent(const bf16_t* __restrict__ ff, const float* __restrict__ Wrec,
              const float* __restrict__ W2, const float* __restrict__ b2,
              float* __restrict__ out) {
  extern __shared__ char smem_raw[];
  bf16_t* sWb  = (bf16_t*)smem_raw;            // 256*264
  bf16_t* sW2  = sWb + 256 * 264;              //  32*264
  bf16_t* sSpk = sW2 + 32 * 264;               //  16*264
  float*  sRed = (float*)(sSpk + 16 * 264);

  const int tid  = threadIdx.x;
  const int lane = tid & 31;
  const int wave = tid >> 5;
  const int half = lane >> 4;
  const int ln   = lane & 15;
  const int b0   = blockIdx.x * 16;

  // Stage weights into LDS (fp32 -> bf16).
  for (int i = tid; i < 256 * 256; i += 256) {
    int n = i >> 8, k = i & 255;
    sWb[n * 264 + k] = (bf16_t)Wrec[i];
  }
  for (int i = tid; i < 32 * 256; i += 256) {
    int n = i >> 8, k = i & 255;
    sW2[n * 264 + k] = (n < N_CLS) ? (bf16_t)W2[n * N_HID + k] : (bf16_t)0.f;
  }
  for (int i = tid; i < 16 * 264; i += 256) sSpk[i] = (bf16_t)0.f;
  if (tid == 0) sRed[0] = 0.f;
  __syncthreads();

  const int nb0 = wave * 32;                   // hidden col bases (2 tiles)
  const int n2  = wave * 16 + ln;              // output col (waves 0,1)
  float b2v = 0.f;
  if (wave < 2 && n2 < N_CLS) b2v = b2[n2];

  v8f mem1[2];
  mem1[0] = (v8f){0.f,0.f,0.f,0.f,0.f,0.f,0.f,0.f};
  mem1[1] = mem1[0];
  v8f mem2 = mem1[0];
  float cnt = 0.f;

  for (int t = 0; t < N_T; ++t) {
    // Feedforward currents for this step (bf16 ws, coalesced per half-wave).
    float ffv[2][8];
#pragma unroll
    for (int tl = 0; tl < 2; ++tl)
#pragma unroll
      for (int r = 0; r < 8; ++r) {
        int m = b0 + r + 8 * half;
        ffv[tl][r] = (float)__builtin_nontemporal_load(
            ff + ((size_t)t * N_B + m) * N_HID + nb0 + tl * 16 + ln);
      }

    // Recurrent matmul: acc = spk1(t-1) @ Wrec^T, K = 256 (8 chained WMMAs).
    v8f acc[2];
    acc[0] = (v8f){0.f,0.f,0.f,0.f,0.f,0.f,0.f,0.f};
    acc[1] = acc[0];
#pragma unroll
    for (int kk = 0; kk < 8; ++kk) {
      // A fragment: two contiguous 8-elem runs (16B-aligned) -> ds_load_b128.
      v16bf A;
      const bf16_t* ar = sSpk + ln * 264 + kk * 32 + 8 * half;
#pragma unroll
      for (int e = 0; e < 8; ++e) {
        A[e]     = ar[e];
        A[8 + e] = ar[16 + e];
      }
#pragma unroll
      for (int tl = 0; tl < 2; ++tl) {
        const bf16_t* br =
            sWb + (nb0 + tl * 16 + ln) * 264 + kk * 32 + 8 * half;
        v16bf Bf;
#pragma unroll
        for (int e = 0; e < 8; ++e) { Bf[e] = br[e]; Bf[8 + e] = br[16 + e]; }   // two 16B runs
        acc[tl] = __builtin_amdgcn_wmma_f32_16x16x32_bf16(
            false, A, false, Bf, (short)0, acc[tl], false, false);
      }
    }
    __syncthreads();   // everyone finished reading spk1(t-1)

    // LIF update for hidden layer; publish spk1(t) to LDS.
#pragma unroll
    for (int tl = 0; tl < 2; ++tl)
#pragma unroll
      for (int r = 0; r < 8; ++r) {
        float m1 = BETA * mem1[tl][r] + ffv[tl][r] + acc[tl][r];
        float sp = (m1 - THRESH > 0.f) ? 1.f : 0.f;
        m1 -= sp * THRESH;
        mem1[tl][r] = m1;
        cnt += sp;
        sSpk[(r + 8 * half) * 264 + nb0 + tl * 16 + ln] = (bf16_t)sp;
      }
    __syncthreads();   // spk1(t) visible to all waves

    // Output layer on waves 0,1: cur2 = spk1(t) @ W2^T (+b2), LIF, emit.
    if (wave < 2) {
      v8f acc2 = (v8f){0.f,0.f,0.f,0.f,0.f,0.f,0.f,0.f};
#pragma unroll
      for (int kk = 0; kk < 8; ++kk) {
        v16bf A;
        const bf16_t* ar = sSpk + ln * 264 + kk * 32 + 8 * half;
#pragma unroll
        for (int e = 0; e < 8; ++e) {
          A[e]     = ar[e];
          A[8 + e] = ar[16 + e];
        }
        const bf16_t* br = sW2 + (wave * 16 + ln) * 264 + kk * 32 + 8 * half;
        v16bf Bf;
#pragma unroll
        for (int e = 0; e < 8; ++e) { Bf[e] = br[e]; Bf[8 + e] = br[16 + e]; }
        acc2 = __builtin_amdgcn_wmma_f32_16x16x32_bf16(
            false, A, false, Bf, (short)0, acc2, false, false);
      }
#pragma unroll
      for (int r = 0; r < 8; ++r) {
        float m2 = BETA * mem2[r] + acc2[r] + b2v;
        float sp = (m2 - THRESH > 0.f) ? 1.f : 0.f;
        m2 -= sp * THRESH;
        mem2[r] = m2;
        if (n2 < N_CLS) {
          cnt += sp;
          int m = b0 + r + 8 * half;
          __builtin_nontemporal_store(
              sp, out + ((size_t)t * N_B + m) * N_CLS + n2);
        }
      }
    }
  }

  // Spike-count reduction: LDS per block, one global atomic per block.
  atomicAdd(sRed, cnt);
  __syncthreads();
  if (tid == 0) atomicAdd(out + OUT_TOTAL_IDX, sRed[0]);
}

__global__ void snn_zero_total(float* __restrict__ out) {
  if (threadIdx.x == 0 && blockIdx.x == 0) out[OUT_TOTAL_IDX] = 0.f;
}

// ---------------------------------------------------------------------------
extern "C" void kernel_launch(void* const* d_in, const int* in_sizes, int n_in,
                              void* d_out, int out_size, void* d_ws, size_t ws_size,
                              hipStream_t stream) {
  const float* x    = (const float*)d_in[0];   // [1024,100,700]
  const float* W1   = (const float*)d_in[1];   // [256,700]
  const float* b1   = (const float*)d_in[2];   // [256]
  const float* Wrec = (const float*)d_in[3];   // [256,256]
  const float* W2   = (const float*)d_in[4];   // [20,256]
  const float* b2   = (const float*)d_in[5];   // [20]
  float* out = (float*)d_out;                  // [100*1024*20] spk_rec + [1] total
  bf16_t* ff = (bf16_t*)d_ws;                  // [100*1024, 256] bf16 = 52.4 MB

  (void)in_sizes; (void)n_in; (void)out_size; (void)ws_size;

  snn_zero_total<<<1, 32, 0, stream>>>(out);

  // Phase 1: 102400 x 256 GEMM, 64-row blocks.
  snn_ff_gemm<<<(N_T * N_B) / 64, 256, 0, stream>>>(x, W1, b1, ff);

  // Phase 2: one block per 16-batch tile, ~157 KB dynamic LDS.
  const size_t smem = (size_t)(256 + 32 + 16) * 264 * sizeof(bf16_t) + 16;
  snn_recurrent<<<N_B / 16, 256, smem, stream>>>(ff, Wrec, W2, b2, out);
}
